// Attention_47493748359836
// MI455X (gfx1250) — matrix operand order repaired and checked by reading, not compiled, not run on hardware
//
#include <hip/hip_runtime.h>
#include <hip/hip_bf16.h>
#include <cstddef>

// ---------------------------------------------------------------------------
// Causal self-attention, B=4, S=2048, D=1024, f32 in/out.
// bf16 inputs + f32 WMMA accumulation (v_wmma_f32_16x16x32_bf16).
//  k0: f32->bf16 convert (x, Wq|Wk|Wv)
//  k1: QKV GEMM, 2x2 register-blocked (32x32 per wave), V stored transposed
//  k2: fused attention: Q tile + 16x2048 f32 score strip in LDS (CDNA5 WGP
//      LDS = 320KB), block softmax, PV with 8-accumulator loop inversion.
// ---------------------------------------------------------------------------

typedef __attribute__((ext_vector_type(16))) __bf16 bf16x16;
typedef __attribute__((ext_vector_type(8)))  __bf16 bf16x8;
typedef __attribute__((ext_vector_type(8)))  float  f32x8;

static constexpr int B  = 4;
static constexpr int S  = 2048;
static constexpr int D  = 1024;
static constexpr int BS = B * S;            // 8192 rows
static constexpr float SCALE = 0.03125f;    // 1/sqrt(1024)

// ---- fragment helpers (contiguous k order, same for A and B) ----------------------

// A-operand (16x32 bf16): lane m<16 holds K 0..15; lane m+16 holds
// K 16..31 of row m. `row` points at K=0 of this lane's row.
__device__ __forceinline__ bf16x16 load_a_frag(const __bf16* row, int hi) {
    const bf16x8 lo = *reinterpret_cast<const bf16x8*>(row + (hi ? 16 : 0));
    const bf16x8 up = *reinterpret_cast<const bf16x8*>(row + (hi ? 24 : 8));
    return __builtin_shufflevector(lo, up, 0,1,2,3,4,5,6,7,8,9,10,11,12,13,14,15);
}

// A-operand built from f32 data (attention probabilities in LDS).
__device__ __forceinline__ bf16x16 cvt_a_frag(const float* row, int hi) {
    const float* p0 = row + (hi ? 16 : 0);
    const float* p1 = row + (hi ? 24 : 8);
    bf16x16 a;
#pragma unroll
    for (int i = 0; i < 8; ++i) {
        a[i]     = (__bf16)p0[i];
        a[i + 8] = (__bf16)p1[i];
    }
    return a;
}

// B-operand (32x16 bf16): lane n<16 holds K=0..15 of column n, lane n+16
// holds K=16..31. `p` points at this lane's first K element (contiguous 16).
__device__ __forceinline__ bf16x16 load_b_frag(const __bf16* p) {
    const bf16x8 lo = *reinterpret_cast<const bf16x8*>(p);
    const bf16x8 up = *reinterpret_cast<const bf16x8*>(p + 8);
    return __builtin_shufflevector(lo, up, 0,1,2,3,4,5,6,7,8,9,10,11,12,13,14,15);
}

__device__ __forceinline__ f32x8 wmma_bf16(bf16x16 a, bf16x16 b, f32x8 c) {
    return __builtin_amdgcn_wmma_f32_16x16x32_bf16(false, a, false, b,
                                                   (short)0, c, false, false);
}

// ---- kernel 0: f32 -> bf16 -------------------------------------------------

__global__ void cvt_f32_bf16(const float* __restrict__ src,
                             __bf16* __restrict__ dst, int n) {
    int i = blockIdx.x * 256 + threadIdx.x;
    if (i < n) dst[i] = (__bf16)src[i];
}

// ---- kernel 1: fused QKV GEMM, 2x2 register blocking ----------------------
// One 32x32 output macro-tile per wave (4 WMMA accumulators). wb = [3][D][D]
// with W[e][d] row-major == B-operand layout for y = x @ W^T.
// mat 0 -> Q [row][e], mat 1 -> K [row][e], mat 2 -> V stored TRANSPOSED
// as vt[b][e][s] so the later PV GEMM gets contiguous B-operand loads.

__global__ void qkv_gemm(const __bf16* __restrict__ xb,
                         const __bf16* __restrict__ wb,
                         __bf16* __restrict__ qb,
                         __bf16* __restrict__ kb,
                         __bf16* __restrict__ vtb) {
    const int tid    = threadIdx.x;
    const int waveId = blockIdx.x * 8 + (tid >> 5);   // jobs: 256*96 = 24576
    const int lane   = tid & 31;
    const int hi     = lane >> 4;
    const int ln     = lane & 15;

    const int mt   = waveId / 96;        // 256 row macro-tiles (32 rows each)
    const int nt   = waveId % 96;        // 96 col macro-tiles (32 cols each)
    const int row0 = mt * 32;
    const int mat  = nt / 32;            // 0=Q 1=K 2=V  (32 macro-cols each)
    const int n0   = (nt % 32) * 32;

    const __bf16* a0p = xb + (size_t)(row0 + ln) * D;
    const __bf16* a1p = xb + (size_t)(row0 + 16 + ln) * D;
    const __bf16* b0p = wb + ((size_t)mat * D + n0 + ln) * D + hi * 16;
    const __bf16* b1p = wb + ((size_t)mat * D + n0 + 16 + ln) * D + hi * 16;

    f32x8 acc00 = {}, acc01 = {}, acc10 = {}, acc11 = {};
#pragma unroll 2
    for (int d = 0; d < D; d += 32) {
        __builtin_prefetch(b0p + d + 256, 0, 1);
        __builtin_prefetch(b1p + d + 256, 0, 1);
        bf16x16 a0 = load_a_frag(a0p + d, hi);
        bf16x16 a1 = load_a_frag(a1p + d, hi);
        bf16x16 b0 = load_b_frag(b0p + d);
        bf16x16 b1 = load_b_frag(b1p + d);
        acc00 = wmma_bf16(a0, b0, acc00);
        acc01 = wmma_bf16(a0, b1, acc01);
        acc10 = wmma_bf16(a1, b0, acc10);
        acc11 = wmma_bf16(a1, b1, acc11);
    }

    const f32x8* accs[4] = { &acc00, &acc01, &acc10, &acc11 };
#pragma unroll
    for (int t = 0; t < 4; ++t) {
        const int mi = t >> 1, ni = t & 1;
        const f32x8& a = *accs[t];
#pragma unroll
        for (int r = 0; r < 8; ++r) {
            const int M    = r + 8 * hi;
            const int grow = row0 + mi * 16 + M;
            const int gcol = n0 + ni * 16 + ln;
            const __bf16 v = (__bf16)a[r];
            if (mat == 0) {
                qb[(size_t)grow * D + gcol] = v;
            } else if (mat == 1) {
                kb[(size_t)grow * D + gcol] = v;
            } else {
                const int bb = grow >> 11;          // /S
                const int s  = grow & (S - 1);
                vtb[((size_t)bb * D + gcol) * S + s] = v;
            }
        }
    }
}

// ---- kernel 2: fused causal attention -------------------------------------
// One block (256 thr = 8 waves) per (batch, 16-query-row tile).
// LDS: Q tile 16x1024 bf16 (32KB) + scores 16x2048 f32 (128KB) + reductions.

static constexpr int Q_ELEMS  = 16 * D;         // bf16
static constexpr int SC_ELEMS = 16 * S;         // f32
static constexpr size_t LDS_BYTES =
    (size_t)Q_ELEMS * 2 + (size_t)SC_ELEMS * 4 + 16 * 16 * 4 + 16 * 4;

__global__ void attn_fused(const __bf16* __restrict__ qb,
                           const __bf16* __restrict__ kb,
                           const __bf16* __restrict__ vtb,
                           float* __restrict__ out) {
    extern __shared__ __align__(16) char smem_raw[];
    __bf16* q_lds = reinterpret_cast<__bf16*>(smem_raw);
    float*  sc    = reinterpret_cast<float*>(smem_raw + (size_t)Q_ELEMS * 2);
    float*  red   = sc + SC_ELEMS;               // [16][16]
    float*  rinv  = red + 16 * 16;               // [16]

    const int tid  = threadIdx.x;
    const int w    = tid >> 5;
    const int lane = tid & 31;
    const int hi   = lane >> 4;
    const int ln   = lane & 15;

    const int bb = blockIdx.x >> 7;              // 128 q-tiles per batch
    const int q0 = (blockIdx.x & 127) * 16;

    const float NINF = -__builtin_inff();
    const int klim   = q0 + 16;                      // causal column bound
    const int klim32 = (klim + 31) & ~31;            // padded to K-step

    // ---- stage Q tile into LDS (16 rows x 1024 bf16 = 2048 uint4) --------
    {
        const uint4* src = reinterpret_cast<const uint4*>(
            qb + (size_t)(bb * S + q0) * D);
        uint4* dst = reinterpret_cast<uint4*>(q_lds);
        for (int i = tid; i < 16 * D / 8; i += 256) dst[i] = src[i];
    }
    // zero the causal padding columns [klim, klim32)
    if (klim32 > klim) {
        const int r = tid & 15, c = klim + (tid >> 4);
        if (c < klim32) sc[r * S + c] = 0.0f;
    }
    __syncthreads();

    // ---- phase 1: scores = scale * Q K^T with causal mask -----------------
    // Each wave handles pairs of adjacent k-tiles: one Q A-frag feeds 2 WMMAs.
    const int ktmax = q0 >> 4;
    const __bf16* arow = q_lds + ln * D;
    for (int p = w; 2 * p <= ktmax; p += 8) {
        const int kt0  = 2 * p;
        const int kt1  = 2 * p + 1;
        const int kt1c = kt1 <= ktmax ? kt1 : ktmax;   // clamp loads in-bounds

        const __bf16* k0base =
            kb + ((size_t)(bb * S + kt0 * 16 + ln)) * D + hi * 16;
        const __bf16* k1base =
            kb + ((size_t)(bb * S + kt1c * 16 + ln)) * D + hi * 16;

        f32x8 acc0 = {}, acc1 = {};
#pragma unroll 2
        for (int d = 0; d < D; d += 32) {
            __builtin_prefetch(k0base + d + 256, 0, 1);
            __builtin_prefetch(k1base + d + 256, 0, 1);
            bf16x16 a  = load_a_frag(arow + d, hi);
            bf16x16 b0 = load_b_frag(k0base + d);
            bf16x16 b1 = load_b_frag(k1base + d);
            acc0 = wmma_bf16(a, b0, acc0);
            acc1 = wmma_bf16(a, b1, acc1);
        }
#pragma unroll
        for (int r = 0; r < 8; ++r) {
            const int M  = r + 8 * hi;
            const int gk = kt0 * 16 + ln;
            float v = acc0[r] * SCALE;
            if (gk > q0 + M) v = NINF;
            sc[M * S + gk] = v;
        }
        if (kt1 <= ktmax) {
#pragma unroll
            for (int r = 0; r < 8; ++r) {
                const int M  = r + 8 * hi;
                const int gk = kt1 * 16 + ln;
                float v = acc1[r] * SCALE;
                if (gk > q0 + M) v = NINF;
                sc[M * S + gk] = v;
            }
        }
    }
    __syncthreads();

    // ---- phase 2: softmax over sc[r][0..klim) (16 threads per row) --------
    {
        const int r = tid & 15, g = tid >> 4;
        float pm = NINF;
        for (int c = g; c < klim; c += 16) pm = fmaxf(pm, sc[r * S + c]);
        red[r * 16 + g] = pm;
        __syncthreads();
        float m = NINF;
#pragma unroll
        for (int i = 0; i < 16; ++i) m = fmaxf(m, red[r * 16 + i]);
        __syncthreads();

        float ps = 0.0f;
        for (int c = g; c < klim; c += 16) {
            const float e = __expf(sc[r * S + c] - m);
            sc[r * S + c] = e;
            ps += e;
        }
        red[r * 16 + g] = ps;
        __syncthreads();
        float sum = 0.0f;
#pragma unroll
        for (int i = 0; i < 16; ++i) sum += red[r * 16 + i];
        if (g == 0) rinv[r] = 1.0f / sum;
    }
    __syncthreads();

    // ---- phase 3: O = P V ; wave w owns output columns [w*128, w*128+128) -
    // Loop inversion: one f32->bf16 P-fragment per K-step feeds 8 n-tiles
    // (8 accumulators); V B-frags fold into immediate offsets off one base.
    {
        const int ncol0 = w * 128;
        const __bf16* vbase =
            vtb + ((size_t)bb * D + ncol0 + ln) * S + hi * 16;
        const float* prow = sc + ln * S;

        f32x8 acc[8] = {};
        for (int k0 = 0; k0 < klim32; k0 += 32) {
            bf16x16 a = cvt_a_frag(prow + k0, hi);
#pragma unroll
            for (int nt = 0; nt < 8; ++nt) {
                bf16x16 b = load_b_frag(vbase + (size_t)nt * 16 * S + k0);
                acc[nt] = wmma_bf16(a, b, acc[nt]);
            }
        }
#pragma unroll
        for (int nt = 0; nt < 8; ++nt) {
#pragma unroll
            for (int r = 0; r < 8; ++r) {
                const int M = r + 8 * hi;
                out[((size_t)bb * S + q0 + M) * D + ncol0 + nt * 16 + ln] =
                    acc[nt][r] * rinv[M];
            }
        }
    }
}

// ---- host glue -------------------------------------------------------------

extern "C" void kernel_launch(void* const* d_in, const int* in_sizes, int n_in,
                              void* d_out, int out_size, void* d_ws, size_t ws_size,
                              hipStream_t stream) {
    const float* x  = (const float*)d_in[0];
    const float* Wq = (const float*)d_in[1];
    const float* Wk = (const float*)d_in[2];
    const float* Wv = (const float*)d_in[3];
    float* out = (float*)d_out;

    constexpr int NX = BS * D;        // 8,388,608
    constexpr int ND = D * D;         // 1,048,576

    __bf16* xb  = (__bf16*)d_ws;
    __bf16* wb  = xb + (size_t)NX;
    __bf16* qb  = wb + (size_t)3 * ND;
    __bf16* kb  = qb + (size_t)NX;
    __bf16* vtb = kb + (size_t)NX;

    // 0) convert inputs to bf16
    cvt_f32_bf16<<<(NX + 255) / 256, 256, 0, stream>>>(x, xb, NX);
    cvt_f32_bf16<<<(ND + 255) / 256, 256, 0, stream>>>(Wq, wb, ND);
    cvt_f32_bf16<<<(ND + 255) / 256, 256, 0, stream>>>(Wk, wb + ND, ND);
    cvt_f32_bf16<<<(ND + 255) / 256, 256, 0, stream>>>(Wv, wb + 2 * ND, ND);

    // 1) QKV projections: 256 row-macrotiles * 96 col-macrotiles, 8/block
    qkv_gemm<<<(256 * 96) / 8, 256, 0, stream>>>(xb, wb, qb, kb, vtb);

    // 2) fused causal attention: one block per (batch, 16-query tile)
    (void)hipFuncSetAttribute(reinterpret_cast<const void*>(attn_fused),
                              hipFuncAttributeMaxDynamicSharedMemorySize,
                              (int)LDS_BYTES);
    attn_fused<<<B * (S / 16), 256, LDS_BYTES, stream>>>(qb, kb, vtb, out);
}
